// MotifEmbedding_2869038154297
// MI455X (gfx1250) — matrix operand order repaired and checked by reading, not compiled, not run on hardware
//
#include <hip/hip_runtime.h>
#include <hip/hip_bf16.h>

// ---- CDNA5 WMMA / TDM vector types (wave32) ----
typedef __attribute__((ext_vector_type(16))) __bf16        v16bf;
typedef __attribute__((ext_vector_type(8)))  float         v8f;
typedef __attribute__((ext_vector_type(4)))  unsigned int  v4u;
typedef __attribute__((ext_vector_type(8)))  int           v8i;
typedef __attribute__((ext_vector_type(4)))  int           v4i;

#define F 128          // feature dim (in = hid = out = 128)

// ---------------- degree / norm kernels ----------------
__global__ __launch_bounds__(256) void k_init_deg(float* deg, int n) {
  int i = blockIdx.x * 256 + threadIdx.x;
  if (i < n) deg[i] = 1.0f;                    // self-loop contributes 1
}

__global__ __launch_bounds__(256) void k_count_deg(float* deg,
                                                   const long long* __restrict__ dst,
                                                   long long E) {
  long long e = (long long)blockIdx.x * 256 + threadIdx.x;
  if (e < E) atomicAdd(&deg[dst[e]], 1.0f);
}

__global__ __launch_bounds__(256) void k_dinv(float* deg, int n) {
  int i = blockIdx.x * 256 + threadIdx.x;
  if (i < n) deg[i] = rsqrtf(deg[i]);          // deg >= 1 always (self-loops)
}

// ---------------- dense lift: H = (relu?)A @ W via WMMA bf16 ----------------
// Block: 256 threads = 8 waves. Wave w computes rows [blk*128 + 16w, +16) x all 128 cols.
// W (128x128 f32) is DMA'd into LDS by the Tensor Data Mover (one wave issues the D#),
// then converted to a transposed bf16 operand buffer: sWT[col*F + k].
template <bool RELU>
__global__ __launch_bounds__(256)
void k_gemm_wmma(const float* __restrict__ A, const float* __restrict__ W,
                 float* __restrict__ H, int n) {
  __shared__ __align__(16) float  sWf32[F * F];  // 64 KB TDM landing buffer
  __shared__ __align__(32) __bf16 sWT[F * F];    // 32 KB bf16 transposed operand

  const int tid = threadIdx.x;

#if __has_builtin(__builtin_amdgcn_tensor_load_to_lds) && \
    __has_builtin(__builtin_amdgcn_s_wait_tensorcnt)
  // ---- TDM: DMA W (16384 dwords, 1-D tensor) global -> LDS ----
  if (tid < 32) {                                // one wave issues; EXEC ignored by TDM
    unsigned long long ga = (unsigned long long)(uintptr_t)W;
    unsigned int ldsOff   = (unsigned int)(uintptr_t)(&sWf32[0]); // LDS byte offset (addr[31:0])
    // D# group 0: count=1 | lds_addr | global_addr[56:0] | type=2
    v4u g0 = { 1u,
               ldsOff,
               (unsigned int)(ga & 0xFFFFFFFFu),
               (unsigned int)((ga >> 32) & 0x01FFFFFFu) | (2u << 30) };
    // D# group 1: data_size=4B (code 2); tensor_dim0 = tile_dim0 = 16384; stride = 16384
    v8i g1 = { (int)(2u << 16),            // [17:16] data_size = 2 -> 4 bytes
               (int)(16384u << 16),        // dword1[31:16] = tensor_dim0[15:0]
               0,                          // tensor_dim0[31:16]=0, tensor_dim1 lo=0
               (int)(16384u << 16),        // dword3[31:16] = tile_dim0 = 16384
               0,                          // tile_dim1=0 (unused), tile_dim2=0
               16384,                      // tensor_dim0_stride[31:0]
               0, 0 };
    v4i gz = {0, 0, 0, 0};
#if defined(__clang_major__) && (__clang_major__ >= 23)
    v8i gz8 = {0, 0, 0, 0, 0, 0, 0, 0};
    __builtin_amdgcn_tensor_load_to_lds(g0, g1, gz, gz, gz8, 0);
#else
    __builtin_amdgcn_tensor_load_to_lds(g0, g1, gz, gz, 0);
#endif
    __builtin_amdgcn_s_wait_tensorcnt(0);        // s_wait_tensorcnt 0
  }
  __syncthreads();
  // Convert + transpose in LDS: sWT[c*F + k] = bf16(W[k][c])
  for (int idx = tid; idx < F * F; idx += 256) {
    int k = idx >> 7, c = idx & 127;
    sWT[c * F + k] = (__bf16)sWf32[idx];
  }
#else
  // Fallback: coalesced global read, strided one-time LDS write
  for (int idx = tid; idx < F * F; idx += 256) {
    int k = idx >> 7, c = idx & 127;
    sWT[c * F + k] = (__bf16)W[idx];
  }
#endif
  __syncthreads();

  const int wave   = tid >> 5;
  const int lane   = tid & 31;
  const int laneLo = lane & 15;      // M (for A/D) or N-col (for B/D)
  const int laneHi = lane >> 4;      // K-half selector
  const int rowBase = blockIdx.x * 128 + wave * 16;

  v8f acc[8] = {};                   // 8 accumulators: 16 x 128 f32 output tile

  // Clamp row so WMMA runs with EXEC all-ones (ISA §7.12 requirement);
  // out-of-range rows are masked only at the store.
  int arow = rowBase + laneLo;
  if (arow >= n) arow = n - 1;
  const float* aptr = A + (size_t)arow * F;

  #pragma unroll
  for (int kt = 0; kt < 4; ++kt) {           // K = 128 in steps of 32
    // A fragment 16x32 bf16 (lanes 0-15 K=0..15, lanes 16-31 K=16..31, matching B)
    v16bf afrag;
    const int k0 = kt * 32 + laneHi * 16;
    #pragma unroll
    for (int i = 0; i < 8; ++i) {
      float v0 = aptr[k0 + i];
      float v1 = aptr[k0 + 8 + i];
      if (RELU) { v0 = fmaxf(v0, 0.0f); v1 = fmaxf(v1, 0.0f); }   // compile-time
      afrag[i]     = (__bf16)v0;
      afrag[i + 8] = (__bf16)v1;
    }
    #pragma unroll
    for (int nt = 0; nt < 8; ++nt) {
      // B fragment 32x16 bf16: lane holds col = laneLo, K = kt*32 + laneHi*16 + {0..15}
      // -> 32 contiguous bytes in LDS (two ds_load_b128)
      const v16bf bfrag =
          *(const v16bf*)&sWT[(nt * 16 + laneLo) * F + kt * 32 + laneHi * 16];
      acc[nt] = __builtin_amdgcn_wmma_f32_16x16x32_bf16(
          /*neg_a=*/false, afrag, /*neg_b=*/false, bfrag,
          /*c_mod=*/(short)0, acc[nt], /*reuse_a=*/false, /*reuse_b=*/false);
    }
  }

  // D layout: VGPR r -> lanes 0-15: M=r, lanes 16-31: M=r+8; N = laneLo
  #pragma unroll
  for (int nt = 0; nt < 8; ++nt) {
    #pragma unroll
    for (int r = 0; r < 8; ++r) {
      int row = rowBase + r + laneHi * 8;
      if (row < n) H[(size_t)row * F + nt * 16 + laneLo] = acc[nt][r];
    }
  }
}

// ---------------- self-loop term + bias (fully overwrites out) ----------------
__global__ __launch_bounds__(256)
void k_selfbias(float* __restrict__ out, const float* __restrict__ h,
                const float* __restrict__ dinv, const float* __restrict__ b,
                long long total) {
  long long i = (long long)blockIdx.x * 256 + threadIdx.x;
  if (i < total) {
    int node = (int)(i >> 7), f = (int)(i & 127);
    float di = dinv[node];
    out[i] = h[i] * di * di + b[f];
  }
}

// ---------------- edge scatter: out[dst] += h[src] * dinv[src]*dinv[dst] ----------------
__global__ __launch_bounds__(256)
void k_scatter(float* __restrict__ out, const float* __restrict__ h,
               const float* __restrict__ dinv,
               const long long* __restrict__ src, const long long* __restrict__ dst,
               long long E) {
  long long t = (long long)blockIdx.x * 256 + threadIdx.x;
  long long e = t >> 7;
  int f = (int)(t & 127);
  if (e < E) {
    long long s = src[e], d = dst[e];
    float nrm = dinv[s] * dinv[d];
    float* op = &out[d * (long long)F + f];
    __builtin_prefetch(op, 1, 0);              // global_prefetch_b8 hint for RMW line
    atomicAdd(op, h[s * (long long)F + f] * nrm);   // fp32 atomic, L2-resident
  }
}

__global__ __launch_bounds__(256) void k_relu(float* x, long long total) {
  long long i = (long long)blockIdx.x * 256 + threadIdx.x;
  if (i < total) x[i] = fmaxf(x[i], 0.0f);
}

// ---------------- driver ----------------
extern "C" void kernel_launch(void* const* d_in, const int* in_sizes, int n_in,
                              void* d_out, int out_size, void* d_ws, size_t ws_size,
                              hipStream_t stream) {
  (void)n_in; (void)out_size; (void)ws_size;
  const float*     x   = (const float*)d_in[0];
  const float*     W1  = (const float*)d_in[1];
  const float*     b1  = (const float*)d_in[2];
  const float*     W2  = (const float*)d_in[3];
  const float*     b2  = (const float*)d_in[4];
  const long long* ei  = (const long long*)d_in[5];

  const int       n   = in_sizes[0] / F;        // 100000
  const long long E   = (long long)in_sizes[5] / 2;
  const long long* src = ei;
  const long long* dst = ei + E;

  // workspace layout: [dinv | h (N*128) | agg (N*128)]
  float* ws   = (float*)d_ws;
  size_t nPad = ((size_t)n + 1023) & ~(size_t)1023;
  float* dinv = ws;
  float* h    = ws + nPad;
  float* agg  = h + (size_t)n * F;
  float* out  = (float*)d_out;

  const long long totNF = (long long)n * F;
  const int nB   = (n + 255) / 256;
  const int eB   = (int)((E + 255) / 256);
  const int gB   = (n + 127) / 128;
  const int nfB  = (int)((totNF + 255) / 256);
  const int efB  = (int)((E * F + 255) / 256);

  // degrees (shared across both layers)
  k_init_deg <<<nB,  256, 0, stream>>>(dinv, n);
  k_count_deg<<<eB,  256, 0, stream>>>(dinv, dst, E);
  k_dinv     <<<nB,  256, 0, stream>>>(dinv, n);

  // layer 1: h1 = x @ W1 ; agg1 = scatter(h1) + b1   (ReLU deferred into layer-2 load)
  k_gemm_wmma<false><<<gB,  256, 0, stream>>>(x, W1, h, n);
  k_selfbias <<<nfB, 256, 0, stream>>>(agg, h, dinv, b1, totNF);
  k_scatter  <<<efB, 256, 0, stream>>>(agg, h, dinv, src, dst, E);

  // layer 2: h2 = relu(agg1) @ W2 ; out = scatter(h2) + b2 ; relu
  k_gemm_wmma<true><<<gB,  256, 0, stream>>>(agg, W2, h, n);
  k_selfbias <<<nfB, 256, 0, stream>>>(out, h, dinv, b2, totNF);
  k_scatter  <<<efB, 256, 0, stream>>>(out, h, dinv, src, dst, E);
  k_relu     <<<nfB, 256, 0, stream>>>(out, totNF);
}
